// MultiRelationGNN_68410239091238
// MI455X (gfx1250) — matrix operand order repaired and checked by reading, not compiled, not run on hardware
//
#include <hip/hip_runtime.h>

#define N_NODES 50000
#define E_EDGES 800000
#define R_REL   3
#define IN_DIM  128
#define HID     128
#define HEADS   4
#define CH      32
#define NEG_SLOPE 0.2f
#define LN_EPS  1e-5f

typedef __attribute__((ext_vector_type(16))) _Float16 v16h;
typedef __attribute__((ext_vector_type(8)))  float    v8f;

// Monotonic float<->uint encoding so float max can use integer atomicMax.
__device__ __forceinline__ unsigned fenc(float f) {
    unsigned u = __float_as_uint(f);
    return (u & 0x80000000u) ? ~u : (u | 0x80000000u);
}
__device__ __forceinline__ float fdec(unsigned u) {
    return (u & 0x80000000u) ? __uint_as_float(u & 0x7FFFFFFFu) : __uint_as_float(~u);
}

__device__ __forceinline__ void gate_softmax(const float* __restrict__ gate,
                                             float& g0, float& g1, float& g2) {
    float a = gate[0], b = gate[1], c = gate[2];
    float m = fmaxf(a, fmaxf(b, c));
    float e0 = __expf(a - m), e1 = __expf(b - m), e2 = __expf(c - m);
    float inv = 1.0f / (e0 + e1 + e2);
    g0 = e0 * inv; g1 = e1 * inv; g2 = e2 * inv;
}

// ---------------------------------------------------------------------------
// K0: zero the accumulator region of d_out, write gw = softmax(gate) at tail.
// ---------------------------------------------------------------------------
__global__ void k_init(float* __restrict__ out, const float* __restrict__ gate) {
    int i = blockIdx.x * blockDim.x + threadIdx.x;
    if (i * 4 < N_NODES * HID) {
        float4 z = make_float4(0.f, 0.f, 0.f, 0.f);
        *(float4*)(out + (size_t)i * 4) = z;
    }
    if (i == 0) {
        float g0, g1, g2;
        gate_softmax(gate, g0, g1, g2);
        out[(size_t)N_NODES * HID + 0] = g0;
        out[(size_t)N_NODES * HID + 1] = g1;
        out[(size_t)N_NODES * HID + 2] = g2;
    }
}

// ---------------------------------------------------------------------------
// K1: h = x @ W_r via v_wmma_f32_16x16x32_f16.
// Block = 256 threads = 8 waves; block owns a 16-row stripe, wave w owns
// output cols [16w, 16w+16). K-loop: 4 WMMA steps of 32.
// ---------------------------------------------------------------------------
__global__ void __launch_bounds__(256) k_gemm(const float* __restrict__ x,
                                              const float* __restrict__ Wall,
                                              float* __restrict__ h, int r) {
    const float* __restrict__ W = Wall + (size_t)r * IN_DIM * HID;
    const int wave = threadIdx.x >> 5;
    const int lane = threadIdx.x & 31;
    const int lm   = lane & 15;
    const int g    = lane >> 4;          // lane group 0/1
    const int m0   = blockIdx.x * 16;
    const int n0   = wave * 16;
    const int m    = m0 + lm;            // A row for this lane
    const int n    = n0 + lm;            // B/C column for this lane

    v8f c = {};
#pragma unroll
    for (int kk = 0; kk < IN_DIM; kk += 32) {
        // A fragment (16x32 f16): lane group g holds K in {g*8..g*8+7} and {16+g*8..16+g*8+7}
        const float* xa = x + (size_t)m * IN_DIM + kk + g * 8;
        float4 x0 = *(const float4*)(xa + 0);
        float4 x1 = *(const float4*)(xa + 4);
        float4 x2 = *(const float4*)(xa + 16);
        float4 x3 = *(const float4*)(xa + 20);
        v16h a;
        a[0]  = (_Float16)x0.x; a[1]  = (_Float16)x0.y; a[2]  = (_Float16)x0.z; a[3]  = (_Float16)x0.w;
        a[4]  = (_Float16)x1.x; a[5]  = (_Float16)x1.y; a[6]  = (_Float16)x1.z; a[7]  = (_Float16)x1.w;
        a[8]  = (_Float16)x2.x; a[9]  = (_Float16)x2.y; a[10] = (_Float16)x2.z; a[11] = (_Float16)x2.w;
        a[12] = (_Float16)x3.x; a[13] = (_Float16)x3.y; a[14] = (_Float16)x3.z; a[15] = (_Float16)x3.w;

        // B fragment (32x16 f16): lane holds column n, K in {g*8..g*8+7} and {16+g*8..16+g*8+7}
        const int kb = kk + g * 8;
        const float* wp = W + (size_t)kb * HID + n;
        v16h b;
#pragma unroll
        for (int j = 0; j < 16; ++j) b[j] = (_Float16)wp[(size_t)(j + (j & 8)) * HID];

        c = __builtin_amdgcn_wmma_f32_16x16x32_f16(false, a, false, b,
                                                   (short)0, c, false, false);
    }

    // C layout: VGPR v -> row m0 + v + 8*g, col n
#pragma unroll
    for (int v = 0; v < 8; ++v) {
        h[(size_t)(m0 + v + 8 * g) * HID + n] = c[v];
    }
}

// ---------------------------------------------------------------------------
// K2: per-(node, head) attention logits; also init emax/denom.
// ---------------------------------------------------------------------------
__global__ void k_alpha(const float* __restrict__ h,
                        const float* __restrict__ att_src_all,
                        const float* __restrict__ att_dst_all,
                        float* __restrict__ a_src, float* __restrict__ a_dst,
                        unsigned* __restrict__ emax, float* __restrict__ denom, int r) {
    int idx = blockIdx.x * blockDim.x + threadIdx.x;
    if (idx >= N_NODES * HEADS) return;
    int node = idx >> 2, hh = idx & 3;
    const float* hp = h + (size_t)node * HID + hh * CH;
    const float* as = att_src_all + (size_t)r * HEADS * CH + hh * CH;
    const float* ad = att_dst_all + (size_t)r * HEADS * CH + hh * CH;
    float ss = 0.f, sd = 0.f;
#pragma unroll
    for (int cc = 0; cc < CH; ++cc) {
        float hv = hp[cc];
        ss += hv * as[cc];
        sd += hv * ad[cc];
    }
    a_src[idx] = ss;
    a_dst[idx] = sd;
    emax[idx]  = 0u;     // smallest encoded value (acts as -inf identity)
    denom[idx] = 0.f;
}

// ---------------------------------------------------------------------------
// K3: segment-max over incoming edges (integer atomicMax on encoded floats).
// ---------------------------------------------------------------------------
__global__ void k_edge_max(const int* __restrict__ ei,
                           const float* __restrict__ a_src,
                           const float* __restrict__ a_dst,
                           unsigned* __restrict__ emax, int r) {
    int e = blockIdx.x * blockDim.x + threadIdx.x;
    if (e >= E_EDGES) return;
    const int* base = ei + (size_t)r * 2 * E_EDGES;
    int s = base[e], d = base[E_EDGES + e];
    float4 as = *(const float4*)(a_src + (size_t)s * 4);
    float4 ad = *(const float4*)(a_dst + (size_t)d * 4);
    float vs[4] = { as.x + ad.x, as.y + ad.y, as.z + ad.z, as.w + ad.w };
#pragma unroll
    for (int hh = 0; hh < 4; ++hh) {
        float v = vs[hh];
        v = v > 0.f ? v : NEG_SLOPE * v;
        atomicMax(&emax[(size_t)d * 4 + hh], fenc(v));
    }
}

// ---------------------------------------------------------------------------
// K4: softmax denominators (atomic f32 add).
// ---------------------------------------------------------------------------
__global__ void k_edge_sum(const int* __restrict__ ei,
                           const float* __restrict__ a_src,
                           const float* __restrict__ a_dst,
                           const unsigned* __restrict__ emax,
                           float* __restrict__ denom, int r) {
    int e = blockIdx.x * blockDim.x + threadIdx.x;
    if (e >= E_EDGES) return;
    const int* base = ei + (size_t)r * 2 * E_EDGES;
    int s = base[e], d = base[E_EDGES + e];
    float4 as = *(const float4*)(a_src + (size_t)s * 4);
    float4 ad = *(const float4*)(a_dst + (size_t)d * 4);
    float vs[4] = { as.x + ad.x, as.y + ad.y, as.z + ad.z, as.w + ad.w };
#pragma unroll
    for (int hh = 0; hh < 4; ++hh) {
        float v = vs[hh];
        v = v > 0.f ? v : NEG_SLOPE * v;
        float m = fdec(emax[(size_t)d * 4 + hh]);
        atomicAdd(&denom[(size_t)d * 4 + hh], __expf(v - m));
    }
}

// ---------------------------------------------------------------------------
// K5: wave-per-edge scatter: out[dst] += gw_r * alpha * h[src].
// Lane handles 4 channels (float4 gather, 4 atomic f32 adds); head = lane/8.
// ---------------------------------------------------------------------------
__global__ void __launch_bounds__(256) k_scatter(const int* __restrict__ ei,
                                                 const float* __restrict__ h,
                                                 const float* __restrict__ a_src,
                                                 const float* __restrict__ a_dst,
                                                 const unsigned* __restrict__ emax,
                                                 const float* __restrict__ denom,
                                                 const float* __restrict__ gate,
                                                 float* __restrict__ out, int r) {
    int gid  = blockIdx.x * blockDim.x + threadIdx.x;
    int e    = gid >> 5;
    if (e >= E_EDGES) return;
    int lane = threadIdx.x & 31;
    const int* base = ei + (size_t)r * 2 * E_EDGES;
    int s = base[e], d = base[E_EDGES + e];
    int hh = lane >> 3;

    float as = a_src[(size_t)s * 4 + hh];
    float ad = a_dst[(size_t)d * 4 + hh];
    float v  = as + ad;
    v = v > 0.f ? v : NEG_SLOPE * v;
    float m    = fdec(emax[(size_t)d * 4 + hh]);
    float den  = denom[(size_t)d * 4 + hh];
    float alph = __expf(v - m) / (den + 1e-16f);

    float g0, g1, g2;
    gate_softmax(gate, g0, g1, g2);
    float gw  = (r == 0) ? g0 : (r == 1) ? g1 : g2;
    float wsc = gw * alph;

    float4 hv = *(const float4*)(h + (size_t)s * HID + lane * 4);
    float* op = out + (size_t)d * HID + lane * 4;
    atomicAdd(op + 0, hv.x * wsc);
    atomicAdd(op + 1, hv.y * wsc);
    atomicAdd(op + 2, hv.z * wsc);
    atomicAdd(op + 3, hv.w * wsc);
}

// ---------------------------------------------------------------------------
// K6: wave-per-node: += sum_r gw_r*bias_r + x, then LayerNorm (wave32 shuffle).
// ---------------------------------------------------------------------------
__global__ void __launch_bounds__(256) k_ln(float* __restrict__ out,
                                            const float* __restrict__ x,
                                            const float* __restrict__ bias,
                                            const float* __restrict__ gate,
                                            const float* __restrict__ gamma,
                                            const float* __restrict__ beta) {
    int gid  = blockIdx.x * blockDim.x + threadIdx.x;
    int node = gid >> 5;
    if (node >= N_NODES) return;
    int lane = threadIdx.x & 31;
    int c0   = lane * 4;

    float g0, g1, g2;
    gate_softmax(gate, g0, g1, g2);

    float4 acc = *(float4*)(out + (size_t)node * HID + c0);
    float4 xr  = *(const float4*)(x + (size_t)node * HID + c0);
    float av[4] = { acc.x, acc.y, acc.z, acc.w };
    float xv[4] = { xr.x,  xr.y,  xr.z,  xr.w  };
    float hv[4];
#pragma unroll
    for (int i = 0; i < 4; ++i) {
        int cc = c0 + i;
        float b = g0 * bias[cc] + g1 * bias[HID + cc] + g2 * bias[2 * HID + cc];
        hv[i] = av[i] + b + xv[i];
    }
    float s  = hv[0] + hv[1] + hv[2] + hv[3];
    float sq = hv[0] * hv[0] + hv[1] * hv[1] + hv[2] * hv[2] + hv[3] * hv[3];
#pragma unroll
    for (int off = 16; off >= 1; off >>= 1) {
        s  += __shfl_xor(s, off, 32);
        sq += __shfl_xor(sq, off, 32);
    }
    float mu  = s * (1.0f / HID);
    float var = sq * (1.0f / HID) - mu * mu;
    float inv = rsqrtf(var + LN_EPS);
    float4 o;
    o.x = (hv[0] - mu) * inv * gamma[c0 + 0] + beta[c0 + 0];
    o.y = (hv[1] - mu) * inv * gamma[c0 + 1] + beta[c0 + 1];
    o.z = (hv[2] - mu) * inv * gamma[c0 + 2] + beta[c0 + 2];
    o.w = (hv[3] - mu) * inv * gamma[c0 + 3] + beta[c0 + 3];
    *(float4*)(out + (size_t)node * HID + c0) = o;
}

// ---------------------------------------------------------------------------
extern "C" void kernel_launch(void* const* d_in, const int* in_sizes, int n_in,
                              void* d_out, int out_size, void* d_ws, size_t ws_size,
                              hipStream_t stream) {
    const float* x       = (const float*)d_in[0];
    const int*   ei      = (const int*)  d_in[1];
    // d_in[2] = edge_attr (unused by GATConv, edge_dim=None)
    const float* W       = (const float*)d_in[3];
    const float* att_src = (const float*)d_in[4];
    const float* att_dst = (const float*)d_in[5];
    const float* bias    = (const float*)d_in[6];
    const float* gate    = (const float*)d_in[7];
    const float* gamma   = (const float*)d_in[8];
    const float* beta    = (const float*)d_in[9];
    float* out = (float*)d_out;

    // Workspace partitioning (reused sequentially across relations): ~28.8 MB
    float*    h      = (float*)d_ws;                         // N*HID
    float*    a_src  = h + (size_t)N_NODES * HID;            // N*HEADS
    float*    a_dst  = a_src + (size_t)N_NODES * HEADS;      // N*HEADS
    unsigned* emax   = (unsigned*)(a_dst + (size_t)N_NODES * HEADS); // N*HEADS
    float*    denom  = (float*)(emax + (size_t)N_NODES * HEADS);     // N*HEADS

    k_init<<<(N_NODES * HID / 4 + 255) / 256, 256, 0, stream>>>(out, gate);

    for (int r = 0; r < R_REL; ++r) {
        k_gemm<<<N_NODES / 16, 256, 0, stream>>>(x, W, h, r);
        k_alpha<<<(N_NODES * HEADS + 255) / 256, 256, 0, stream>>>(
            h, att_src, att_dst, a_src, a_dst, emax, denom, r);
        k_edge_max<<<(E_EDGES + 255) / 256, 256, 0, stream>>>(ei, a_src, a_dst, emax, r);
        k_edge_sum<<<(E_EDGES + 255) / 256, 256, 0, stream>>>(ei, a_src, a_dst, emax, denom, r);
        k_scatter<<<((size_t)E_EDGES * 32 + 255) / 256, 256, 0, stream>>>(
            ei, h, a_src, a_dst, emax, denom, gate, out, r);
    }

    k_ln<<<(N_NODES * 32 + 255) / 256, 256, 0, stream>>>(out, x, bias, gate, gamma, beta);
}
